// VAE_RNN_narli_52261162058632
// MI455X (gfx1250) — matrix operand order repaired and checked by reading, not compiled, not run on hardware
//
#include <hip/hip_runtime.h>

// ---------------------------------------------------------------------------
// VRNN (variational RNN) forward, MI455X / gfx1250.
// Strategy: batch-parallel persistent workgroups (32 WGs x 16 batch rows),
// sequential over T=1024 inside the kernel. All GEMMs via
// v_wmma_f32_16x16x32_bf16 (f32 accum). GRU weights + head weights resident
// in LDS (loaded once with the Tensor Data Mover); square-layer weights
// streamed from L2 each step (L2-hot, 192MB). fp32 master copy of h for
// recurrence stability.
// ---------------------------------------------------------------------------

typedef __bf16 bf16_t;
typedef bf16_t v16bf __attribute__((ext_vector_type(16)));
typedef bf16_t v8bf  __attribute__((ext_vector_type(8)));
typedef float  v8f   __attribute__((ext_vector_type(8)));
typedef unsigned int v4u __attribute__((ext_vector_type(4)));
typedef int    v8i   __attribute__((ext_vector_type(8)));
typedef int    v4i   __attribute__((ext_vector_type(4)));

#define T_LEN 1024
#define NBATCH 512
#define LOG2PI 1.8378770664093453f

// ---- packed bf16 weight image layout in d_ws (bytes) ----------------------
// resident region [0, RESIDENT_BYTES) is copied verbatim to LDS by the TDM.
#define OFF_WIH       0u          // [K=128][N=384] 24 ntiles x 4 kchunks
#define OFF_WHH       98304u
#define OFF_PHI_Z1    196608u     // K=32  N=128
#define OFF_ENC_MEAN  204800u     // K=128 N=32
#define OFF_ENC_LV    212992u
#define OFF_PR_MEAN   221184u
#define OFF_PR_LV     229376u
#define OFF_DEC_MEAN  237568u     // K=128 N=16 (8 valid)
#define OFF_DEC_LV    241664u
#define OFF_BIAS      245760u     // f32 bias slots (see B_* below)
#define RESIDENT_BYTES 253952u    // 248KB, multiple of 8
// streamed region (read from global each step, L2-resident)
#define OFF_PHI_Y     253952u     // K=32(8 valid) N=128
#define OFF_PHI_U     262144u
#define OFF_ENC       270336u     // K=256 N=128
#define OFF_PRIOR1    335872u     // K=128 N=128
#define OFF_PRIOR2    368640u
#define OFF_PHI_Z2    401408u
#define OFF_DEC       434176u
#define OFF_PARTIALS  466944u     // 32 f32 block partial losses
// bias slots (float index within bias region)
#define B_PHI_Y 0
#define B_PHI_U 128
#define B_PHI_Z1 256
#define B_PHI_Z2 384
#define B_ENC 512
#define B_ENC_MEAN 640
#define B_ENC_LV 672
#define B_PRIOR1 704
#define B_PRIOR2 832
#define B_PR_MEAN 960
#define B_PR_LV 992
#define B_DEC 1024
#define B_DEC_MEAN 1152
#define B_DEC_LV 1168

// ---- LDS layout (bytes). Resident weight image mirrored at [0,RESIDENT). --
#define LDS_PYH   253952u   // [16][256] bf16  (cols 0-127 = phi_y(y_t), 128-255 = h)
#define LDS_PU    262144u   // [16][128] bf16
#define LDS_PR1   266240u
#define LDS_ENCH  270336u
#define LDS_PR2   274432u
#define LDS_Z     278528u   // [16][32] bf16
#define LDS_PZ1   279552u
#define LDS_PZ2   283648u
#define LDS_DECH  287744u
#define LDS_SEMU  291840u   // [16][32] f32 stat buffers
#define LDS_SELV  293888u
#define LDS_SPMU  295936u
#define LDS_SPLV  297984u
#define LDS_SDMU  300032u   // [16][16] f32
#define LDS_SDLV  301056u
#define LDS_RG    302080u   // [16][128] bf16 GRU r gate
#define LDS_ZG    306176u   // [16][128] bf16 GRU z gate
#define LDS_H     310272u   // [16][128] f32 master hidden state
#define LDS_YT    318464u   // [16][8] f32 y_t for loss
#define LDS_AYT   318976u   // [16][32] bf16 y_t (A operand, cols>=8 zero)
#define LDS_AUT   320000u
#define LDS_RED   321024u   // 256 f32 reduction scratch
#define LDS_BYTES 322048u

// ---------------------------------------------------------------------------
__device__ __forceinline__ v8f wmma_bf16(v16bf a, v16bf b, v8f c) {
  return __builtin_amdgcn_wmma_f32_16x16x32_bf16(false, a, false, b, (short)0, c,
                                                 false, false);
}

// A fragment (16x32 bf16) from row-major [16][ld] activation buffer.
// A layout: lane->row m=lane&15; k = (lane>=16?8:0) + (e<8 ? e : 16+e-8).
// => two contiguous 8-element (16B) row segments.
__device__ __forceinline__ v16bf fragA(const bf16_t* act, int ld, int lane, int kcol0) {
  const int m  = lane & 15;
  const int kb = kcol0 + ((lane >> 4) << 3);
  v8bf lo = *(const v8bf*)(act + m * ld + kb);
  v8bf hi = *(const v8bf*)(act + m * ld + kb + 16);
  v16bf r;
#pragma unroll
  for (int i = 0; i < 8; ++i) { r[i] = lo[i]; r[i + 8] = hi[i]; }
  return r;
}

// B fragment (32x16 bf16) from fragment-packed storage: 32B contiguous/lane.
__device__ __forceinline__ v16bf fragB(const bf16_t* frag, int lane) {
  return *(const v16bf*)(frag + lane * 16);
}

// D tile (f32 16x16 acc) -> bf16 row-major [16][ld], +bias, optional relu.
// D layout: n = lane&15, m = v + (lane>=16 ? 8 : 0).
__device__ __forceinline__ void storeTileBf16(bf16_t* dst, int ld, int lane, v8f d,
                                              const float* bias16, bool relu) {
  const int n = lane & 15, mb = (lane >> 4) << 3;
  const float b = bias16[n];
#pragma unroll
  for (int v = 0; v < 8; ++v) {
    float x = d[v] + b;
    if (relu) x = fmaxf(x, 0.f);
    dst[(mb + v) * ld + n] = (bf16_t)x;
  }
}

__device__ __forceinline__ void storeTileF32(float* dst, int ld, int lane, v8f d,
                                             const float* bias16, bool relu) {
  const int n = lane & 15, mb = (lane >> 4) << 3;
  const float b = bias16[n];
#pragma unroll
  for (int v = 0; v < 8; ++v) {
    float x = d[v] + b;
    if (relu) x = fmaxf(x, 0.f);
    dst[(mb + v) * ld + n] = x;
  }
}

// ---------------------------------------------------------------------------
// Prep: repack all fp32 weights into fragment-packed bf16 image + f32 biases.
// B element mapping: within a fragment, element (lane,e): n = lane&15,
// klocal = (lane>=16 ? 8 : 0) + (e < 8 ? e : 8 + e).
// ---------------------------------------------------------------------------
struct WPtrs {
  const float* w[16];
  const float* b[14];
};

__global__ void prep_weights(WPtrs p, unsigned char* ws) {
  const int KREAL[16] = {128,128, 32,128,128,128,128,128,128,  8,  8,256,128,128,128,128};
  const int NREAL[16] = {384,384,128, 32, 32, 32, 32,  8,  8,128,128,128,128,128,128,128};
  const int KC[16]    = {  4,  4,  1,  4,  4,  4,  4,  4,  4,  1,  1,  8,  4,  4,  4,  4};
  const int FR[16]    = { 96, 96,  8,  8,  8,  8,  8,  4,  4,  8,  8, 64, 32, 32, 32, 32};
  const unsigned OFF[16] = {OFF_WIH, OFF_WHH, OFF_PHI_Z1, OFF_ENC_MEAN, OFF_ENC_LV,
                            OFF_PR_MEAN, OFF_PR_LV, OFF_DEC_MEAN, OFF_DEC_LV,
                            OFF_PHI_Y, OFF_PHI_U, OFF_ENC, OFF_PRIOR1, OFF_PRIOR2,
                            OFF_PHI_Z2, OFF_DEC};
  const int stride = gridDim.x * blockDim.x;
  const int idx0 = blockIdx.x * blockDim.x + threadIdx.x;

  // weights: 448 fragments x 512 bf16 elements
  for (int i = idx0; i < 448 * 512; i += stride) {
    int fg = i >> 9, within = i & 511, lane = within >> 4, e = within & 15;
    int layer = 0, base = 0;
    while (fg >= base + FR[layer]) { base += FR[layer]; ++layer; }
    int fr = fg - base;
    int kcCount = KC[layer];
    int nt = fr / kcCount, kc = fr % kcCount;
    int k = kc * 32 + ((lane >> 4) << 3) + (e < 8 ? e : 8 + e);
    int n = nt * 16 + (lane & 15);
    float v = 0.f;
    if (k < KREAL[layer] && n < NREAL[layer]) v = p.w[layer][n * KREAL[layer] + k];
    ((bf16_t*)(ws + OFF[layer]))[(unsigned)fr * 512u + (unsigned)within] = (bf16_t)v;
  }
  // biases (padded slots, rest zero)
  const int BCNT[14] = {128,128,128,128,128, 32, 32,128,128, 32, 32,128,  8,  8};
  const int BOFF[14] = {  0,128,256,384,512,640,672,704,832,960,992,1024,1152,1168};
  float* bdst = (float*)(ws + OFF_BIAS);
  for (int i = idx0; i < 2048; i += stride) {
    float v = 0.f;
#pragma unroll
    for (int j = 0; j < 14; ++j)
      if (i >= BOFF[j] && i < BOFF[j] + BCNT[j]) v = p.b[j][i - BOFF[j]];
    bdst[i] = v;
  }
}

// ---------------------------------------------------------------------------
// Main persistent kernel: one workgroup per 16 batch rows, loop over T.
// 256 threads = 8 wave32s; wave w owns N-tile(s) w of each layer.
// ---------------------------------------------------------------------------
__global__ __launch_bounds__(256) void vrnn_kernel(
    const float* __restrict__ uu, const float* __restrict__ yy,
    const float* __restrict__ eps, const unsigned char* __restrict__ wimg,
    float* __restrict__ partials) {
  extern __shared__ char smem[];
  const int tid = threadIdx.x, lane = tid & 31, wv = tid >> 5;
  const int bg0 = blockIdx.x * 16;

  // ---- LDS pointers -------------------------------------------------------
  const bf16_t* LW = (const bf16_t*)smem;          // resident weight image
  const float*  LB = (const float*)(smem + OFF_BIAS);
  bf16_t* PYH  = (bf16_t*)(smem + LDS_PYH);
  bf16_t* APU  = (bf16_t*)(smem + LDS_PU);
  bf16_t* APR1 = (bf16_t*)(smem + LDS_PR1);
  bf16_t* AENCH= (bf16_t*)(smem + LDS_ENCH);
  bf16_t* APR2 = (bf16_t*)(smem + LDS_PR2);
  bf16_t* AZ   = (bf16_t*)(smem + LDS_Z);
  bf16_t* APZ1 = (bf16_t*)(smem + LDS_PZ1);
  bf16_t* APZ2 = (bf16_t*)(smem + LDS_PZ2);
  bf16_t* ADECH= (bf16_t*)(smem + LDS_DECH);
  float*  SEMU = (float*)(smem + LDS_SEMU);
  float*  SELV = (float*)(smem + LDS_SELV);
  float*  SPMU = (float*)(smem + LDS_SPMU);
  float*  SPLV = (float*)(smem + LDS_SPLV);
  float*  SDMU = (float*)(smem + LDS_SDMU);
  float*  SDLV = (float*)(smem + LDS_SDLV);
  bf16_t* RG   = (bf16_t*)(smem + LDS_RG);
  bf16_t* ZGB  = (bf16_t*)(smem + LDS_ZG);
  float*  HF   = (float*)(smem + LDS_H);
  float*  YT   = (float*)(smem + LDS_YT);
  bf16_t* AYT  = (bf16_t*)(smem + LDS_AYT);
  bf16_t* AUT  = (bf16_t*)(smem + LDS_AUT);
  float*  RED  = (float*)(smem + LDS_RED);

  // streamed global weight fragments (bf16 elements)
  const bf16_t* GW      = (const bf16_t*)wimg;
  const bf16_t* WPHIY   = GW + OFF_PHI_Y / 2;
  const bf16_t* WPHIU   = GW + OFF_PHI_U / 2;
  const bf16_t* WENC    = GW + OFF_ENC / 2;
  const bf16_t* WPR1    = GW + OFF_PRIOR1 / 2;
  const bf16_t* WPR2    = GW + OFF_PRIOR2 / 2;
  const bf16_t* WPZ2    = GW + OFF_PHI_Z2 / 2;
  const bf16_t* WDEC    = GW + OFF_DEC / 2;
  // LDS-resident fragments
  const bf16_t* LWIH    = LW + OFF_WIH / 2;
  const bf16_t* LWHH    = LW + OFF_WHH / 2;
  const bf16_t* LPZ1W   = LW + OFF_PHI_Z1 / 2;
  const bf16_t* LEMUW   = LW + OFF_ENC_MEAN / 2;
  const bf16_t* LELVW   = LW + OFF_ENC_LV / 2;
  const bf16_t* LPMUW   = LW + OFF_PR_MEAN / 2;
  const bf16_t* LPLVW   = LW + OFF_PR_LV / 2;
  const bf16_t* LDMUW   = LW + OFF_DEC_MEAN / 2;
  const bf16_t* LDLVW   = LW + OFF_DEC_LV / 2;

  // ---- TDM: DMA the resident weight+bias image (248KB) global -> LDS -----
  if (wv == 0) {
    unsigned long long ga = (unsigned long long)wimg;
    const unsigned td0 = RESIDENT_BYTES / 8u;   // qwords, fits tile_dim0 (16b)
    v4u g0 = {1u,                                // count=1, user descriptor
              0u,                                // lds_addr = 0
              (unsigned)ga,
              (unsigned)((ga >> 32) & 0x01FFFFFFu) | 0x80000000u}; // type=2
    v8i g1 = {(int)(3u << 16),                   // data_size = 8B
              (int)((td0 & 0xFFFFu) << 16),      // tensor_dim0 lo
              (int)((td0 >> 16) | (1u << 16)),   // tensor_dim0 hi | tensor_dim1=1
              (int)(td0 << 16),                  // tile_dim0 = td0
              1,                                 // tile_dim1=1, tile_dim2=0
              (int)td0,                          // tensor_dim0_stride lo
              0, 0};
    v4i gz4 = {0, 0, 0, 0};
    v8i gz8 = {0, 0, 0, 0, 0, 0, 0, 0};
    __builtin_amdgcn_tensor_load_to_lds(g0, g1, gz4, gz4, gz8, 0);
    __builtin_amdgcn_s_wait_tensorcnt(0);
  }
  // ---- init h = 0, zero padded K columns of y/u A tiles -------------------
  for (int i = tid; i < 16 * 128; i += 256) {
    HF[i] = 0.f;
    PYH[(i >> 7) * 256 + 128 + (i & 127)] = (bf16_t)0.f;
  }
  for (int i = tid; i < 16 * 32; i += 256) {
    if ((i & 31) >= 8) { AYT[i] = (bf16_t)0.f; AUT[i] = (bf16_t)0.f; }
  }
  __syncthreads();

  float lacc = 0.f;

  for (int t = 0; t < T_LEN; ++t) {
    // P0: stage y_t, u_t tiles ([16][8], strided gathers, L2-hot over t)
    if (tid < 128) {
      const int m = tid >> 3, c = tid & 7;
      const size_t base = ((size_t)(bg0 + m) * 8 + c) * T_LEN + t;
      const float yv = yy[base], uv = uu[base];
      YT[m * 8 + c] = yv;
      AYT[m * 32 + c] = (bf16_t)yv;
      AUT[m * 32 + c] = (bf16_t)uv;
      if (t + 1 < T_LEN) {
        __builtin_prefetch(&yy[base + 1], 0, 0);
        __builtin_prefetch(&uu[base + 1], 0, 0);
      }
    }
    __syncthreads();

    // P1: py = phi_y(y_t), pu = phi_u(u_t), pr1 = relu(prior1(h))
    {
      v8f a = {};
      a = wmma_bf16(fragA(AYT, 32, lane, 0), fragB(WPHIY + wv * 512, lane), a);
      storeTileBf16(PYH + wv * 16, 256, lane, a, LB + B_PHI_Y + wv * 16, false);
      v8f b = {};
      b = wmma_bf16(fragA(AUT, 32, lane, 0), fragB(WPHIU + wv * 512, lane), b);
      storeTileBf16(APU + wv * 16, 128, lane, b, LB + B_PHI_U + wv * 16, false);
      v8f c = {};
#pragma unroll
      for (int kc = 0; kc < 4; ++kc)
        c = wmma_bf16(fragA(PYH, 256, lane, 128 + kc * 32),
                      fragB(WPR1 + (wv * 4 + kc) * 512, lane), c);
      storeTileBf16(APR1 + wv * 16, 128, lane, c, LB + B_PRIOR1 + wv * 16, true);
    }
    __syncthreads();

    // P2: enc_h = relu(enc([py,h])) (K=256), pr2 = relu(prior2(pr1))
    {
      v8f a = {};
#pragma unroll
      for (int kc = 0; kc < 8; ++kc)
        a = wmma_bf16(fragA(PYH, 256, lane, kc * 32),
                      fragB(WENC + (wv * 8 + kc) * 512, lane), a);
      storeTileBf16(AENCH + wv * 16, 128, lane, a, LB + B_ENC + wv * 16, true);
      v8f b = {};
#pragma unroll
      for (int kc = 0; kc < 4; ++kc)
        b = wmma_bf16(fragA(APR1, 128, lane, kc * 32),
                      fragB(WPR2 + (wv * 4 + kc) * 512, lane), b);
      storeTileBf16(APR2 + wv * 16, 128, lane, b, LB + B_PRIOR2 + wv * 16, true);
    }
    __syncthreads();

    // P3: four stat heads [16,128]x[128,32]; wave pairs own each head
    {
      const int nt = wv & 1, which = wv >> 1;
      const bf16_t* asrc = (which < 2) ? AENCH : APR2;
      const bf16_t* bw   = (which == 0) ? LEMUW : (which == 1) ? LELVW
                                        : (which == 2) ? LPMUW : LPLVW;
      const float* bias  = LB + ((which == 0) ? B_ENC_MEAN : (which == 1) ? B_ENC_LV
                                : (which == 2) ? B_PR_MEAN : B_PR_LV);
      float* dst = (which == 0) ? SEMU : (which == 1) ? SELV
                 : (which == 2) ? SPMU : SPLV;
      const bool relu = (which == 1) || (which == 3);
      v8f a = {};
#pragma unroll
      for (int kc = 0; kc < 4; ++kc)
        a = wmma_bf16(fragA(asrc, 128, lane, kc * 32),
                      fragB(bw + (nt * 4 + kc) * 512, lane), a);
      storeTileF32(dst + nt * 16, 32, lane, a, bias + nt * 16, relu);
    }
    __syncthreads();

    // P4: z = mu + exp(.5 lv) * eps ; accumulate KLD
    for (int i = tid; i < 512; i += 256) {
      const int m = i >> 5, zc = i & 31;
      const float emu = SEMU[i], elv = SELV[i], pmu = SPMU[i], plv = SPLV[i];
      const float ep = eps[((size_t)t * NBATCH + bg0 + m) * 32 + zc];
      AZ[i] = (bf16_t)(emu + __expf(0.5f * elv) * ep);
      const float dmu = emu - pmu;
      lacc += 0.5f * (plv - elv - 1.f + (__expf(elv) + dmu * dmu) * __expf(-plv));
    }
    __syncthreads();

    // P5: pz1 = relu(phi_z1(z)); GRU r & z gates (gx + gh accumulated)
    {
      v8f a = {};
      a = wmma_bf16(fragA(AZ, 32, lane, 0), fragB(LPZ1W + wv * 512, lane), a);
      storeTileBf16(APZ1 + wv * 16, 128, lane, a, LB + B_PHI_Z1 + wv * 16, true);

      v8f r = {}, zg = {};
#pragma unroll
      for (int kc = 0; kc < 4; ++kc) {
        v16bf apu = fragA(APU, 128, lane, kc * 32);
        v16bf ah  = fragA(PYH, 256, lane, 128 + kc * 32);
        r  = wmma_bf16(apu, fragB(LWIH + ((0 * 8 + wv) * 4 + kc) * 512, lane), r);
        r  = wmma_bf16(ah,  fragB(LWHH + ((0 * 8 + wv) * 4 + kc) * 512, lane), r);
        zg = wmma_bf16(apu, fragB(LWIH + ((8 + wv) * 4 + kc) * 512, lane), zg);
        zg = wmma_bf16(ah,  fragB(LWHH + ((8 + wv) * 4 + kc) * 512, lane), zg);
      }
      const int n = lane & 15, mb = (lane >> 4) << 3;
#pragma unroll
      for (int v = 0; v < 8; ++v) {
        RG[(mb + v) * 128 + wv * 16 + n]  = (bf16_t)(1.f / (1.f + __expf(-r[v])));
        ZGB[(mb + v) * 128 + wv * 16 + n] = (bf16_t)(1.f / (1.f + __expf(-zg[v])));
      }
    }
    __syncthreads();

    // P6a: pz2 = phi_z2(pz1); GRU n-gate partials (gxn, ghn separate)
    v8f gxn = {}, ghn = {};
    {
      v8f a = {};
#pragma unroll
      for (int kc = 0; kc < 4; ++kc) {
        a   = wmma_bf16(fragA(APZ1, 128, lane, kc * 32),
                        fragB(WPZ2 + (wv * 4 + kc) * 512, lane), a);
        gxn = wmma_bf16(fragA(APU, 128, lane, kc * 32),
                        fragB(LWIH + ((16 + wv) * 4 + kc) * 512, lane), gxn);
        ghn = wmma_bf16(fragA(PYH, 256, lane, 128 + kc * 32),
                        fragB(LWHH + ((16 + wv) * 4 + kc) * 512, lane), ghn);
      }
      storeTileBf16(APZ2 + wv * 16, 128, lane, a, LB + B_PHI_Z2 + wv * 16, false);
    }
    __syncthreads();

    // P6b: n = tanh(gxn + r*ghn); h = (1-z)*n + z*h (fp32 master state)
    {
      const int n = lane & 15, mb = (lane >> 4) << 3, ng = wv * 16 + n;
#pragma unroll
      for (int v = 0; v < 8; ++v) {
        const int m = mb + v;
        const float rr = (float)RG[m * 128 + ng];
        const float zz = (float)ZGB[m * 128 + ng];
        const float nn = tanhf(gxn[v] + rr * ghn[v]);
        const float hn = (1.f - zz) * nn + zz * HF[m * 128 + ng];
        HF[m * 128 + ng] = hn;
        PYH[m * 256 + 128 + ng] = (bf16_t)hn;
      }
    }
    __syncthreads();

    // P7: dec_h = relu(dec(pz2))
    {
      v8f a = {};
#pragma unroll
      for (int kc = 0; kc < 4; ++kc)
        a = wmma_bf16(fragA(APZ2, 128, lane, kc * 32),
                      fragB(WDEC + (wv * 4 + kc) * 512, lane), a);
      storeTileBf16(ADECH + wv * 16, 128, lane, a, LB + B_DEC + wv * 16, true);
    }
    __syncthreads();

    // P8: dec_mean / dec_logvar heads (waves 0,1)
    if (wv < 2) {
      const bool isLv = (wv == 1);
      v8f a = {};
      const bf16_t* bw = isLv ? LDLVW : LDMUW;
#pragma unroll
      for (int kc = 0; kc < 4; ++kc)
        a = wmma_bf16(fragA(ADECH, 128, lane, kc * 32), fragB(bw + kc * 512, lane), a);
      storeTileF32(isLv ? SDLV : SDMU, 16, lane, a,
                   LB + (isLv ? B_DEC_LV : B_DEC_MEAN), isLv);
    }
    __syncthreads();

    // P9: gaussian NLL (loss += -logp)
    if (tid < 128) {
      const int m = tid >> 3, c = tid & 7;
      const float mu = SDMU[m * 16 + c], lv = SDLV[m * 16 + c];
      const float d = YT[m * 8 + c] - mu;
      lacc += 0.5f * d * d * __expf(-lv) + 0.5f * lv + 0.5f * LOG2PI;
    }
    __syncthreads();
  }

  // block reduction -> per-block partial (deterministic final sum elsewhere)
  RED[tid] = lacc;
  __syncthreads();
  for (int s = 128; s > 0; s >>= 1) {
    if (tid < s) RED[tid] += RED[tid + s];
    __syncthreads();
  }
  if (tid == 0) partials[blockIdx.x] = RED[0];
}

__global__ void finalize_kernel(const float* __restrict__ partials,
                                float* __restrict__ out, int nb) {
  if (threadIdx.x == 0) {
    float s = 0.f;
    for (int i = 0; i < nb; ++i) s += partials[i];  // fixed order: deterministic
    out[0] = s;
  }
}

// ---------------------------------------------------------------------------
extern "C" void kernel_launch(void* const* d_in, const int* in_sizes, int n_in,
                              void* d_out, int out_size, void* d_ws, size_t ws_size,
                              hipStream_t stream) {
  (void)in_sizes; (void)n_in; (void)out_size; (void)ws_size;
  // setup_inputs() order: u(0), y(1), eps(2), then params in insertion order,
  // each (W,b) pair flattened W then b; gru weights have no bias.
  WPtrs p;
  const int widx[16] = {31, 32, 7, 13, 15, 21, 23, 27, 29, 3, 5, 11, 17, 19, 9, 25};
  const int bidx[14] = {4, 6, 8, 10, 12, 14, 16, 18, 20, 22, 24, 26, 28, 30};
  for (int i = 0; i < 16; ++i) p.w[i] = (const float*)d_in[widx[i]];
  for (int i = 0; i < 14; ++i) p.b[i] = (const float*)d_in[bidx[i]];

  unsigned char* ws = (unsigned char*)d_ws;
  prep_weights<<<512, 256, 0, stream>>>(p, ws);
  vrnn_kernel<<<NBATCH / 16, 256, LDS_BYTES, stream>>>(
      (const float*)d_in[0], (const float*)d_in[1], (const float*)d_in[2], ws,
      (float*)(ws + OFF_PARTIALS));
  finalize_kernel<<<1, 32, 0, stream>>>((const float*)(ws + OFF_PARTIALS),
                                        (float*)d_out, NBATCH / 16);
}
